// SO3EquivConvBlock_47510928228405
// MI455X (gfx1250) — matrix operand order repaired and checked by reading, not compiled, not run on hardware
//
#include <hip/hip_runtime.h>
#include <math.h>

// ---------------------------------------------------------------------------
// SO3EquivConvBlock, restructured for MI455X (gfx1250, wave32, WMMA).
//   conv[g,o] = (1/k)[ sum_{c,i} w3[c,o*64+i]*M[g,c,i] + sum_i b3[o*64+i]*S[g,i] ]
//   M[g,c,i]  = sum_k h2[g,k,c]*F[g,k,i],   S[g,i] = sum_k F[g,k,i]
// Three kernels: w3 repack (f32->f16, [o][ci] layout), per-point MLP + per-group
// M/S (WMMA), and the 2048x8256x64 group GEMM (WMMA).  Mflat (34MB) + w3c (1MB)
// live in d_ws and stay resident in the 192MB L2.
// Epilogues are branchless (A&S erf w/ v_rcp+v_exp, hw sin/cos) so TRANS32 work
// co-executes with the WMMA pipeline instead of serializing on EXEC-mask saves.
// ---------------------------------------------------------------------------

typedef _Float16 f16;
typedef _Float16 v8h  __attribute__((ext_vector_type(8)));
typedef _Float16 v16h __attribute__((ext_vector_type(16)));
typedef float    v8f  __attribute__((ext_vector_type(8)));

#define NPTS    65536
#define KPG     32
#define NGRP    2048        // NPTS / KPG
#define IN_DIM  64
#define OUT_DIM 64
#define FDIM    16
#define HDIM    128
#define KCI     8256        // HDIM*IN_DIM + IN_DIM (b3*S folded as extra channels)
#define KCHUNK  258         // KCI / 32
#define PI_FREQ 50.26548245743669f   // pi * 16

static_assert(KCI == HDIM * IN_DIM + IN_DIM, "K extension mismatch");

__device__ __forceinline__ v8f wmma_f16(v16h a, v16h b, v8f c) {
  // v_wmma_f32_16x16x32_f16 (8-arg form: neg_a, A, neg_b, B, c_mod, C, reuse_a, reuse_b)
  return __builtin_amdgcn_wmma_f32_16x16x32_f16(false, a, false, b, (short)0, c,
                                                false, false);
}

// Gather two contiguous 8-half runs into one 16-half WMMA operand.
__device__ __forceinline__ v16h pack16(const f16* lo, const f16* hi) {
  v8h a = *(const v8h*)lo;
  v8h b = *(const v8h*)hi;
  v16h r;
#pragma unroll
  for (int i = 0; i < 8; ++i) { r[i] = a[i]; r[i + 8] = b[i]; }
  return r;
}

// Branchless erf (Abramowitz & Stegun 7.1.26, |err| < ~1.5e-7 + rcp error) —
// far below f16 activation quantization.  v_rcp_f32 + v_exp_f32 are TRANS32
// ops that co-execute with WMMA; no IEEE div sequence, no EXEC divergence.
__device__ __forceinline__ float erf_fast(float x) {
  float ax = fabsf(x);
  float t  = __builtin_amdgcn_rcpf(fmaf(0.3275911f, ax, 1.0f));  // denom >= 1
  float p  = t * fmaf(t, fmaf(t, fmaf(t, fmaf(t, 1.061405429f, -1.453152027f),
                                      1.421413741f),
                              -0.284496736f),
                      0.254829592f);
  float e  = 1.0f - p * __expf(-ax * ax);
  return copysignf(e, x);
}

__device__ __forceinline__ float gelu_fast(float x) {
  return 0.5f * x * (1.0f + erf_fast(x * 0.70710678118654752440f));
}

// ---------------------------------------------------------------------------
// Kernel R: repack w3 (f32 [c][o*64+i]) + b3 into f16 w3c[o][ci], ci = c*64+i,
// with b3 appended at ci in [8192, 8256).  B-tiles then load contiguously.
// ---------------------------------------------------------------------------
__global__ void repack_w3(const float* __restrict__ w3, const float* __restrict__ b3,
                          f16* __restrict__ w3c) {
  int t = blockIdx.x * blockDim.x + threadIdx.x;
  if (t >= OUT_DIM * KCI) return;
  int o = t / KCI, ci = t % KCI;
  float v;
  if (ci < HDIM * IN_DIM) {
    int c = ci >> 6, i = ci & 63;
    v = w3[(size_t)c * (OUT_DIM * IN_DIM) + o * IN_DIM + i];
  } else {
    v = b3[o * IN_DIM + (ci - HDIM * IN_DIM)];
  }
  w3c[(size_t)o * KCI + ci] = (f16)v;
}

// ---------------------------------------------------------------------------
// Kernel 1: one wave per 32-point group (lane == point).
//   geometry -> theta -> fourier feats -> gelu MLP (WMMA) -> M = h2^T @ F (WMMA)
// Writes Mflat[g][0..8191] = M, Mflat[g][8192..8255] = S (f16).
// ---------------------------------------------------------------------------
#define WPB1 4
__global__ __launch_bounds__(32 * WPB1) void point_group_kernel(
    const float* __restrict__ X, const float* __restrict__ F,
    const float* __restrict__ Wf, const float* __restrict__ w1,
    const float* __restrict__ b1, const float* __restrict__ w2,
    const float* __restrict__ b2, f16* __restrict__ Mflat) {
  // Weights transposed for contiguous B-operand runs: [n][k].
  __shared__ __align__(16) f16 w1t[HDIM * 32];
  __shared__ __align__(16) f16 w2t[HDIM * HDIM];
  __shared__ float b1s[HDIM], b2s[HDIM];
  __shared__ __align__(16) f16 sfeat[WPB1][KPG * 32];    // [p][j] fourier feats
  __shared__ __align__(16) f16 h1t[WPB1][KPG * HDIM];    // [p][c]
  __shared__ __align__(16) f16 h2tT[WPB1][HDIM * KPG];   // [c][p] (transposed for M stage A-tiles)

  const int tid  = threadIdx.x;
  const int wave = tid >> 5;
  const int lane = tid & 31;
  const int g    = blockIdx.x * WPB1 + wave;

  // Cooperative weight stage (f32 -> f16, transposed).
  for (int idx = tid; idx < 32 * HDIM; idx += blockDim.x) {
    int k = idx / HDIM, n = idx % HDIM;
    w1t[n * 32 + k] = (f16)w1[idx];
  }
  for (int idx = tid; idx < HDIM * HDIM; idx += blockDim.x) {
    int k = idx / HDIM, n = idx % HDIM;
    w2t[n * HDIM + k] = (f16)w2[idx];
  }
  for (int idx = tid; idx < HDIM; idx += blockDim.x) {
    b1s[idx] = b1[idx];
    b2s[idx] = b2[idx];
  }
  __syncthreads();

  // --- geometry: rotate group so q (lane 0's point) -> north pole; theta ---
  {
    const float* xp = X + (size_t)(g * KPG + lane) * 3;
    float x0 = xp[0], x1 = xp[1], x2 = xp[2];
    float q0 = __shfl(x0, 0, 32);
    float q1 = __shfl(x1, 0, 32);
    float q2 = __shfl(x2, 0, 32);
    // v = cross(q, [0,0,1]) = (q1, -q0, 0)
    float v0 = q1, v1 = -q0;
    float c1x = v1 * x2, c1y = -v0 * x2, c1z = v0 * x1 - v1 * x0;
    float c2z = v0 * c1y - v1 * c1x;
    float rotz = x2 + c1z + c2z / (1.0f + q2 + 1e-8f);
    rotz = fminf(fmaxf(rotz, -1.0f + 1e-6f), 1.0f - 1e-6f);
    float theta = acosf(rotz);
#pragma unroll
    for (int j = 0; j < FDIM; ++j) {
      float pr = theta * (Wf[j] * PI_FREQ);   // |pr| <~ 3200: hw sin/cos fine
      sfeat[wave][lane * 32 + j]        = (f16)__sinf(pr);
      sfeat[wave][lane * 32 + FDIM + j] = (f16)__cosf(pr);
    }
  }
  __syncthreads();

  const int lm     = lane & 15;
  const int hiHalf = (lane >= 16);
  const int k0lo   = hiHalf ? 16 : 0;  // A-operand low run  {0..7} / {16..23}
  const int k0hi   = k0lo + 8;         // A-operand high run {8..15} / {24..31}
  const int kb     = hiHalf ? 16 : 0;  // B-operand run      {0..15} / {16..31}

  // --- layer 1: h1 = gelu(feat @ w1 + b1), K = 32 exactly one WMMA ---
  v16h aF0 = pack16(&sfeat[wave][(0 * 16 + lm) * 32 + k0lo],
                    &sfeat[wave][(0 * 16 + lm) * 32 + k0hi]);
  v16h aF1 = pack16(&sfeat[wave][(1 * 16 + lm) * 32 + k0lo],
                    &sfeat[wave][(1 * 16 + lm) * 32 + k0hi]);
#pragma unroll
  for (int nt = 0; nt < 8; ++nt) {
    int n = nt * 16 + lm;
    v16h b = pack16(&w1t[n * 32 + kb], &w1t[n * 32 + kb + 8]);
    v8f acc0 = {}, acc1 = {};
    acc0 = wmma_f16(aF0, b, acc0);
    acc1 = wmma_f16(aF1, b, acc1);
    float bias = b1s[n];
#pragma unroll
    for (int r = 0; r < 8; ++r) {
      int p0 = r + (hiHalf ? 8 : 0);
      h1t[wave][p0 * HDIM + n]        = (f16)gelu_fast(acc0[r] + bias);
      h1t[wave][(16 + p0) * HDIM + n] = (f16)gelu_fast(acc1[r] + bias);
    }
  }
  __syncthreads();

  // --- layer 2: h2 = gelu(h1 @ w2 + b2), K = 128 (4 WMMA K-steps) ---
  v16h a2[2][4];
#pragma unroll
  for (int mt = 0; mt < 2; ++mt)
#pragma unroll
    for (int kc = 0; kc < 4; ++kc) {
      const f16* base = &h1t[wave][(mt * 16 + lm) * HDIM + kc * 32];
      a2[mt][kc] = pack16(base + k0lo, base + k0hi);
    }
#pragma unroll
  for (int nt = 0; nt < 8; ++nt) {
    int n = nt * 16 + lm;
    v8f acc0 = {}, acc1 = {};
#pragma unroll
    for (int kc = 0; kc < 4; ++kc) {
      const f16* bb = &w2t[n * HDIM + kc * 32 + kb];
      v16h b = pack16(bb, bb + 8);
      acc0 = wmma_f16(a2[0][kc], b, acc0);
      acc1 = wmma_f16(a2[1][kc], b, acc1);
    }
    float bias = b2s[n];
#pragma unroll
    for (int r = 0; r < 8; ++r) {
      int p0 = r + (hiHalf ? 8 : 0);
      h2tT[wave][n * KPG + p0]      = (f16)gelu_fast(acc0[r] + bias);  // [c][p]
      h2tT[wave][n * KPG + 16 + p0] = (f16)gelu_fast(acc1[r] + bias);
    }
  }
  __syncthreads();

  // --- M = h2^T @ F (contract over the 32 points), plus S = sum_k F ---
  const float* Fg = F + (size_t)g * KPG * IN_DIM;
  v16h bF[4];
  float ssum[4];
#pragma unroll
  for (int it = 0; it < 4; ++it) {
    int i = it * 16 + lm;
    float s = 0.0f;
    v16h t;
#pragma unroll
    for (int kk = 0; kk < 16; ++kk) {
      float fv = Fg[(kb + kk) * IN_DIM + i];
      s += fv;
      t[kk] = (f16)fv;
    }
    bF[it] = t;
    s += __shfl_xor(s, 16, 32);  // combine complementary k-halves
    ssum[it] = s;
  }

  f16* Mg = Mflat + (size_t)g * KCI;
#pragma unroll
  for (int ct = 0; ct < 8; ++ct) {
    int c = ct * 16 + lm;
    const f16* ab = &h2tT[wave][c * KPG];
    v16h aC = pack16(ab + k0lo, ab + k0hi);
#pragma unroll
    for (int it = 0; it < 4; ++it) {
      v8f m = {};
      m = wmma_f16(aC, bF[it], m);
#pragma unroll
      for (int r = 0; r < 8; ++r) {
        int crow = ct * 16 + r + (hiHalf ? 8 : 0);
        Mg[crow * IN_DIM + it * 16 + lm] = (f16)m[r];
      }
    }
  }
  if (lane < 16) {  // predicated stores only (all WMMAs above run with full EXEC)
#pragma unroll
    for (int it = 0; it < 4; ++it)
      Mg[HDIM * IN_DIM + it * 16 + lane] = (f16)ssum[it];
  }
}

// ---------------------------------------------------------------------------
// Kernel 2: out[2048][64] = (1/32) * Mflat[2048][8256] @ w3c^T.
// One wave per 16x16 output tile: 512 waves, 258 K-chunks each.
// Operands streamed from L2-resident workspace; prefetch ahead.
// ---------------------------------------------------------------------------
#define WPB2 8
__global__ __launch_bounds__(32 * WPB2) void group_gemm_kernel(
    const f16* __restrict__ Mflat, const f16* __restrict__ w3c,
    float* __restrict__ out) {
  const int tid  = threadIdx.x;
  const int wave = tid >> 5;
  const int lane = tid & 31;
  const int w    = blockIdx.x * WPB2 + wave;
  const int gt   = w >> 2;  // group tile [0,128)
  const int ot   = w & 3;   // out tile   [0,4)
  const int lm     = lane & 15;
  const int hiHalf = (lane >= 16);
  const int k0lo   = hiHalf ? 16 : 0, k0hi = k0lo + 8;
  const int kb     = hiHalf ? 16 : 0;

  const f16* arow = Mflat + (size_t)(gt * 16 + lm) * KCI;
  const f16* brow = w3c + (size_t)(ot * 16 + lm) * KCI;
  v8f acc = {};
  for (int kc = 0; kc < KCHUNK; ++kc) {
    int kbase = kc * 32;
    __builtin_prefetch(arow + kbase + 512, 0, 1);  // global_prefetch_b8
    __builtin_prefetch(brow + kbase + 512, 0, 1);
    v16h a = pack16(arow + kbase + k0lo, arow + kbase + k0hi);
    v16h b = pack16(brow + kbase + kb, brow + kbase + kb + 8);
    acc = wmma_f16(a, b, acc);
  }
  const float inv_k = 1.0f / (float)KPG;
#pragma unroll
  for (int r = 0; r < 8; ++r) {
    int gg = gt * 16 + r + (hiHalf ? 8 : 0);
    out[(size_t)gg * OUT_DIM + ot * 16 + lm] = acc[r] * inv_k;
  }
}

// ---------------------------------------------------------------------------
extern "C" void kernel_launch(void* const* d_in, const int* in_sizes, int n_in,
                              void* d_out, int out_size, void* d_ws, size_t ws_size,
                              hipStream_t stream) {
  (void)in_sizes; (void)n_in; (void)out_size; (void)ws_size;
  const float* X  = (const float*)d_in[0];
  const float* F  = (const float*)d_in[1];
  const float* Wf = (const float*)d_in[2];
  const float* w1 = (const float*)d_in[3];
  const float* b1 = (const float*)d_in[4];
  const float* w2 = (const float*)d_in[5];
  const float* b2 = (const float*)d_in[6];
  const float* w3 = (const float*)d_in[7];
  const float* b3 = (const float*)d_in[8];
  float* out = (float*)d_out;

  // Workspace: Mflat 2048*8256*2 = 33,816,576 B, then w3c 64*8256*2 = 1,056,768 B.
  f16* Mflat = (f16*)d_ws;
  f16* w3c   = (f16*)((char*)d_ws + (size_t)NGRP * KCI * sizeof(f16));

  repack_w3<<<(OUT_DIM * KCI + 255) / 256, 256, 0, stream>>>(w3, b3, w3c);
  point_group_kernel<<<NGRP / WPB1, 32 * WPB1, 0, stream>>>(X, F, Wf, w1, b1, w2,
                                                            b2, Mflat);
  group_gemm_kernel<<<(NGRP / 16) * 4 / WPB2, 32 * WPB2, 0, stream>>>(Mflat, w3c,
                                                                      out);
}
